// TransConv_48077863912017
// MI455X (gfx1250) — matrix operand order repaired and checked by reading, not compiled, not run on hardware
//
#include <hip/hip_runtime.h>

// ---------------------------------------------------------------------------
// TransConv (stride-2, 4x4 kernel) as implicit GEMM on CDNA5 WMMA (bf16).
//   out[n,co,oy,ox] = b[co] + sum_{ci,ky,kx valid} W[ky,kx,ci,co]*x[n,ci,iy,ix]
// kernel==2*stride => exactly 2x2 taps per output pixel:
//   ky = 2*tyi + (oy&1), iy = (oy + (oy&1) - 2)/2 + tyi
//   kx = 2*txi + p (p = ox&1, ox = 2j+p), ix = j + p + txi - 1
// Per (n, oy, p, co-half): GEMM  M=64(j) x N=64(co) x K=512(4 taps * 128 ci).
// ---------------------------------------------------------------------------

typedef __attribute__((ext_vector_type(16))) __bf16 v16bf;
typedef __attribute__((ext_vector_type(8)))  float  v8f;

#define CI    128
#define CO    128
#define IN_H  64
#define IN_W  64
#define OUT_H 128
#define OUT_W 128

// LDS layout (ushort units, all 8B-aligned):
//  x region : [2 rows][66 ixp][132 ci-padded]   -> A-frag K contiguous in ci
//  W region : [8 taps][64 co][132 ci-padded]    -> B-frag K contiguous in ci
#define XROW   132
#define XREG   (2 * 66 * XROW)        // 17424 ushorts = 34848 B
#define WOFF   XREG                   // ushort offset of W region
#define WCOL   132
#define WREG   (8 * 64 * WCOL)        // 67584 ushorts = 135168 B
#define SMEM_BYTES ((XREG + WREG) * 2)  // 170016 B (< 320KB WGP LDS)

struct F16Frag { union { v16bf v; unsigned long long q[4]; }; };

__device__ __forceinline__ unsigned short f2bf(float f) {
  unsigned u = __builtin_bit_cast(unsigned, f);
  u += 0x7FFFu + ((u >> 16) & 1u);    // round-to-nearest-even
  return (unsigned short)(u >> 16);
}

__global__ __launch_bounds__(256)
void tconv_wmma_kernel(const float* __restrict__ x, const float* __restrict__ W,
                       const float* __restrict__ bias, float* __restrict__ out) {
  extern __shared__ unsigned short smem[];

  const int oy  = blockIdx.x;           // 0..127
  const int n   = blockIdx.y;           // 0..31
  const int co0 = blockIdx.z * 64;      // co half
  const int tid  = threadIdx.x;
  const int lane = tid & 31;
  const int wave = tid >> 5;            // 8 wave32s
  const int p    = wave >> 2;           // ox parity owned by this wave
  const int mt   = wave & 3;            // 16-row M tile within j=0..63

  const int oyp    = oy & 1;
  const int iyBase = (oy + oyp - 2) >> 1;   // iy of tap tyi=0 (may be -1)

  // ---- zero x region (covers conv boundary guards + invalid iy rows) ----
  {
    unsigned int* z = (unsigned int*)smem;
    for (int i = tid; i < (XREG >> 1); i += 256) z[i] = 0u;
  }
  __syncthreads();

  // ---- stage x -> LDS bf16, transposed to [row][ixp][ci] (ixp = ix+1) ----
  for (int e = tid; e < 2 * CI * IN_W; e += 256) {     // 16384 elems
    const int ix  = e & 63;
    const int ci  = (e >> 6) & 127;
    const int row = e >> 13;                           // tyi
    const int iy  = iyBase + row;
    if (iy >= 0 && iy < IN_H) {
      const float v = x[((n * CI + ci) * IN_H + iy) * IN_W + ix];
      smem[(row * 66 + ix + 1) * XROW + ci] = f2bf(v);
    }
  }

  // ---- stage W -> LDS bf16, transposed to [tyi*4+kx][co][ci] ----
  for (int e = tid; e < 65536; e += 256) {
    const int col = e & 63;
    const int ci  = (e >> 6) & 127;
    const int kx  = (e >> 13) & 3;
    const int tyi = e >> 15;
    const int ky  = 2 * tyi + oyp;
    const float v = W[((ky * 4 + kx) * CI + ci) * CO + co0 + col];
    smem[WOFF + ((tyi * 4 + kx) * 64 + col) * WCOL + ci] = f2bf(v);
  }
  __syncthreads();

  // ---- implicit GEMM: M=64(j, this parity) x N=64(co-half) x K=512 ----
  v8f acc0 = {}, acc1 = {}, acc2 = {}, acc3 = {};
  const int j    = mt * 16 + (lane & 15);   // output pixel row of A
  const int kbA  = (lane >> 4) * 16;        // A lane K-base, contiguous k = 16*(lane/16) + s
  const int kbB  = (lane >> 4) * 16;        // B lane K-base, same contiguous k order as A
  const int colB = lane & 15;               // N column within tile
  const unsigned long long* q64 = (const unsigned long long*)smem;

#pragma unroll
  for (int s = 0; s < 16; ++s) {            // 16 K-steps of 32
    const int t   = s >> 2;                 // tap 0..3
    const int cs  = (s & 3) * 32;           // ci base
    const int tyi = t >> 1;
    const int txi = t & 1;
    const int tt  = tyi * 4 + 2 * txi + p;  // LDS tap slot (kx = 2*txi+p)

    // A fragment: lane K elements = kbA + 0..15, contiguous in ci
    F16Frag A;
    {
      const int ab = ((tyi * 66 + j + p + txi) * XROW + cs + kbA) >> 2;
      A.q[0] = q64[ab];      A.q[1] = q64[ab + 1];
      A.q[2] = q64[ab + 2];  A.q[3] = q64[ab + 3];
    }
    // B fragments: lane = N column, K = kbB + 0..15 contiguous in ci
    F16Frag B0, B1, B2, B3;
    {
      const int bb = (WOFF + (tt * 64 + colB) * WCOL + cs + kbB) >> 2;
      const int ns = (16 * WCOL) >> 2;      // N-tile stride in u64 = 528
      B0.q[0] = q64[bb + 0 * ns];     B0.q[1] = q64[bb + 0 * ns + 1];
      B0.q[2] = q64[bb + 0 * ns + 2]; B0.q[3] = q64[bb + 0 * ns + 3];
      B1.q[0] = q64[bb + 1 * ns];     B1.q[1] = q64[bb + 1 * ns + 1];
      B1.q[2] = q64[bb + 1 * ns + 2]; B1.q[3] = q64[bb + 1 * ns + 3];
      B2.q[0] = q64[bb + 2 * ns];     B2.q[1] = q64[bb + 2 * ns + 1];
      B2.q[2] = q64[bb + 2 * ns + 2]; B2.q[3] = q64[bb + 2 * ns + 3];
      B3.q[0] = q64[bb + 3 * ns];     B3.q[1] = q64[bb + 3 * ns + 1];
      B3.q[2] = q64[bb + 3 * ns + 2]; B3.q[3] = q64[bb + 3 * ns + 3];
    }
    acc0 = __builtin_amdgcn_wmma_f32_16x16x32_bf16(false, A.v, false, B0.v,
                                                   (short)0, acc0, false, false);
    acc1 = __builtin_amdgcn_wmma_f32_16x16x32_bf16(false, A.v, false, B1.v,
                                                   (short)0, acc1, false, false);
    acc2 = __builtin_amdgcn_wmma_f32_16x16x32_bf16(false, A.v, false, B2.v,
                                                   (short)0, acc2, false, false);
    acc3 = __builtin_amdgcn_wmma_f32_16x16x32_bf16(false, A.v, false, B3.v,
                                                   (short)0, acc3, false, false);
  }

  __syncthreads();   // x region no longer needed; reuse as f32 staging

  // ---- accumulators -> LDS staging [co 64][ox 128] (re-interleave parity) --
  float* stag = (float*)smem;               // 64*128*4 = 32768 B <= x region
  const int rhi = (lane >> 4) * 8;          // D-matrix: lanes>=16 hold M=r+8
#pragma unroll
  for (int r = 0; r < 8; ++r) {
    const int m  = mt * 16 + r + rhi;       // == j
    const int ox = 2 * m + p;
    stag[(colB)      * OUT_W + ox] = acc0[r];
    stag[(16 + colB) * OUT_W + ox] = acc1[r];
    stag[(32 + colB) * OUT_W + ox] = acc2[r];
    stag[(48 + colB) * OUT_W + ox] = acc3[r];
  }
  __syncthreads();

  // ---- bias + fully coalesced global stores (512B per co row) ----
  for (int idx = tid; idx < 64 * OUT_W; idx += 256) {
    const int col = idx >> 7;
    const int ox  = idx & 127;
    const float v = stag[idx] + bias[co0 + col];
    out[((n * CO + co0 + col) * OUT_H + oy) * OUT_W + ox] = v;
  }
}

extern "C" void kernel_launch(void* const* d_in, const int* in_sizes, int n_in,
                              void* d_out, int out_size, void* d_ws, size_t ws_size,
                              hipStream_t stream) {
  (void)in_sizes; (void)n_in; (void)out_size; (void)d_ws; (void)ws_size;
  const float* x = (const float*)d_in[0];
  const float* W = (const float*)d_in[1];
  const float* b = (const float*)d_in[2];
  float* out = (float*)d_out;

  dim3 grid(OUT_H, 32, 2);   // (oy, n, co-half)
  dim3 block(256);           // 8 wave32
  tconv_wmma_kernel<<<grid, block, SMEM_BYTES, stream>>>(x, W, b, out);
}
